// VectorQuantizer_18932215840813
// MI455X (gfx1250) — matrix operand order repaired and checked by reading, not compiled, not run on hardware
//
#include <hip/hip_runtime.h>
#include <cstdint>
#include <cstddef>

// ---------------------------------------------------------------------------
// VQ-VAE vector quantizer for MI455X (gfx1250, wave32, WMMA).
// sim GEMM: [32768,256] x [256,1024] via v_wmma_f32_16x16x32_bf16 with
// hi/lo bf16 split (3 WMMAs per tile) for ~fp32 accuracy.
// B tiles staged in LDS (shared by all 8 waves) via async-to-LDS copies,
// double buffered, tracked with ASYNCcnt.
// ---------------------------------------------------------------------------

typedef __attribute__((ext_vector_type(16))) __bf16    v16bf;
typedef __attribute__((ext_vector_type(8)))  float     v8f;
typedef __attribute__((ext_vector_type(4)))  uint32_t  u32x4;
typedef __attribute__((ext_vector_type(4)))  int       i32x4;

union V16BF {
  v16bf    v;
  uint32_t d[8];
  u32x4    q[2];
};

#define D_DIM   256      // embedding dim (GEMM inner dim)
#define K_CODES 1024     // number of codebook entries
#define N_ROWS  32768    // 32*32*32 flattened rows
#define NT      64       // N-tiles of 16 codes
#define NCHUNK  8        // K-chunks of 32 over D=256

// ws layout (bytes)
#define BHI_OFF  0u                      // 512 KB packed bf16-hi B image
#define BLO_OFF  524288u                 // 512 KB packed bf16-lo B image
#define ET_OFF   1048576u                // 1 MB transposed codebook f32
#define E2_OFF   2097152u                // 4 KB column norms
#define PART_OFF 2101248u                // 1 KB per-block loss partials

#if __has_builtin(__builtin_amdgcn_global_load_async_to_lds_b128)
#define USE_ASYNC_LDS 1
#else
#define USE_ASYNC_LDS 0
#endif

#define GLOBAL_AS __attribute__((address_space(1)))
#define LDS_AS    __attribute__((address_space(3)))

// round-to-nearest-even f32 -> bf16, packed pair into one dword
static __device__ __forceinline__ uint32_t f2bf_pair(float f0, float f1) {
  uint32_t u0 = __builtin_bit_cast(uint32_t, f0);
  uint32_t u1 = __builtin_bit_cast(uint32_t, f1);
  uint32_t r0 = (u0 + (0x7FFFu + ((u0 >> 16) & 1u))) >> 16;
  uint32_t r1 = (u1 + (0x7FFFu + ((u1 >> 16) & 1u))) >> 16;
  return (r1 << 16) | (r0 & 0xFFFFu);
}
static __device__ __forceinline__ float bf_lo_as_f32(uint32_t pair) {
  return __builtin_bit_cast(float, pair << 16);
}
static __device__ __forceinline__ float bf_hi_as_f32(uint32_t pair) {
  return __builtin_bit_cast(float, pair & 0xFFFF0000u);
}

// ---------------------------------------------------------------------------
// Prep 1: eT[k][d] = emb[d][k] (coalesced gather later) and e2[k] = ||e_k||^2
// ---------------------------------------------------------------------------
__global__ void vq_prep_e(const float* __restrict__ emb,
                          float* __restrict__ eT,
                          float* __restrict__ e2) {
  int k = blockIdx.x * 256 + threadIdx.x;   // 0..1023
  float s = 0.f;
#pragma unroll 4
  for (int d = 0; d < D_DIM; ++d) {
    float v = emb[(size_t)d * K_CODES + k]; // coalesced across k
    s += v * v;
    eT[(size_t)k * D_DIM + d] = v;
  }
  e2[k] = s;
}

// ---------------------------------------------------------------------------
// Prep 2: pre-pack codebook into per-lane B-matrix (32x16 bf16) VGPR images.
// ISA 16-bit B layout: lane = 16*h + n ; vgpr v packs K = 16*(v>>2) + 8h + 2*(v&3) + {0,1}.
// Storage: dword index = ((t*8 + c)*32 + lane)*8 + v  (lane-major -> b128 loads)
// ---------------------------------------------------------------------------
__global__ void vq_pack_b(const float* __restrict__ emb,
                          uint32_t* __restrict__ bhi,
                          uint32_t* __restrict__ blo) {
  const int t    = blockIdx.x;          // 0..63
  const int c    = threadIdx.x >> 5;    // 0..7
  const int lane = threadIdx.x & 31;
  const int kcode = t * 16 + (lane & 15);
  const int dbase = c * 32 + ((lane >> 4) << 3);

  V16BF hv, lv;
#pragma unroll
  for (int v = 0; v < 8; ++v) {
    float f0 = emb[(size_t)(dbase + ((v >> 2) << 4) + 2 * (v & 3) + 0) * K_CODES + kcode];
    float f1 = emb[(size_t)(dbase + ((v >> 2) << 4) + 2 * (v & 3) + 1) * K_CODES + kcode];
    uint32_t hp = f2bf_pair(f0, f1);
    float h0 = bf_lo_as_f32(hp), h1 = bf_hi_as_f32(hp);
    uint32_t lp = f2bf_pair(f0 - h0, f1 - h1);
    hv.d[v] = hp;
    lv.d[v] = lp;
  }
  size_t base = (((size_t)t * 8 + c) * 32 + lane) * 8;
  *(u32x4*)(bhi + base)     = hv.q[0];
  *(u32x4*)(bhi + base + 4) = hv.q[1];
  *(u32x4*)(blo + base)     = lv.q[0];
  *(u32x4*)(blo + base + 4) = lv.q[1];
}

#if USE_ASYNC_LDS
static __device__ __forceinline__ void async_cp16(const uint32_t* gsrc,
                                                  uint32_t* lds_dst) {
  __builtin_amdgcn_global_load_async_to_lds_b128(
      (GLOBAL_AS i32x4*)gsrc, (LDS_AS i32x4*)lds_dst, 0, 0);
}
// copy one N-tile's packed B image (8 KB hi + 8 KB lo) into an LDS buffer;
// 256 threads x 4 async B128 each = 16 KB, ASYNCcnt += 4 per thread
static __device__ __forceinline__ void fill_buf_async(
    const uint32_t* __restrict__ bhi, const uint32_t* __restrict__ blo,
    uint32_t* sbuf, int t, int tid) {
#pragma unroll
  for (int p = 0; p < 2; ++p) {
    int dw = (tid + p * 256) * 4;                 // dword 0..2047
    async_cp16(bhi + (size_t)t * 2048 + dw, sbuf + dw);
    async_cp16(blo + (size_t)t * 2048 + dw, sbuf + 2048 + dw);
  }
}
#endif

// ---------------------------------------------------------------------------
// Main: fused sim-GEMM + argmin + gather + loss partial.
// 256 blocks x 256 threads (8 waves). Wave w owns rows [blk*128 + w*16, +16).
// ---------------------------------------------------------------------------
__global__ void __launch_bounds__(256)
vq_main(const float* __restrict__ x,
        const float* __restrict__ eT,
        const float* __restrict__ e2,
        const uint32_t* __restrict__ bhi,
        const uint32_t* __restrict__ blo,
        float* __restrict__ out,
        float* __restrict__ partial) {
  const int lane    = threadIdx.x & 31;
  const int wave    = threadIdx.x >> 5;
  const int rowbase = blockIdx.x * 128 + wave * 16;
  const int myrow   = rowbase + (lane & 15);
  const int kb_half = (lane >> 4) << 3;   // 0 or 8 (A-matrix K sub-block)

  // ---- Build resident A tile (16x256) as hi/lo bf16 register images ----
  // ISA 16-bit A layout: lane m = lane&15 ; K(v,s) = 16*(v>>2) + kb_half + 2*(v&3) + s
  V16BF a_hi[NCHUNK], a_lo[NCHUNK];
  const float* xrow = x + (size_t)myrow * D_DIM;
#pragma unroll
  for (int c = 0; c < NCHUNK; ++c) {
#pragma unroll
    for (int v = 0; v < 8; ++v) {
      int kbase = ((v >> 2) << 4) + kb_half + ((v & 3) << 1);
      const float2 f2 = *(const float2*)(xrow + c * 32 + kbase);
      uint32_t hp = f2bf_pair(f2.x, f2.y);
      float h0 = bf_lo_as_f32(hp), h1 = bf_hi_as_f32(hp);
      uint32_t lp = f2bf_pair(f2.x - h0, f2.y - h1);
      a_hi[c].d[v] = hp;
      a_lo[c].d[v] = lp;
    }
  }

  float mins[8];
  int   mini[8];
#pragma unroll
  for (int v = 0; v < 8; ++v) { mins[v] = __builtin_inff(); mini[v] = 0; }

#if USE_ASYNC_LDS
  // double-buffered LDS staging of B tiles (16 KB per buffer)
  __shared__ uint32_t smem[2][4096];
  const int tid = threadIdx.x;
  fill_buf_async(bhi, blo, &smem[0][0], 0, tid);   // prologue: buf0 <- tile 0
#endif

  // ---- Sweep all 64 N-tiles of 16 codes ----
  for (int t = 0; t < NT; ++t) {
#if USE_ASYNC_LDS
    __syncthreads();   // all waves done reading the buffer we overwrite next
    if (t + 1 < NT) {
      fill_buf_async(bhi, blo, &smem[(t + 1) & 1][0], t + 1, tid);
      // 4 new ops in flight; wait for the previous tile's 4 to complete
      asm volatile("s_wait_asynccnt 0x4" ::: "memory");
    } else {
      asm volatile("s_wait_asynccnt 0x0" ::: "memory");
    }
    __syncthreads();   // every wave's piece of the current buffer has landed
    const uint32_t* sb = &smem[t & 1][0];
#endif
    v8f acc0 = {}, acc1 = {}, acc2 = {};   // 3 independent WMMA chains
#pragma unroll
    for (int c = 0; c < NCHUNK; ++c) {
      V16BF bh, bl;
#if USE_ASYNC_LDS
      int idx = (c * 32 + lane) * 8;
      bh.q[0] = *(const u32x4*)(sb + idx);
      bh.q[1] = *(const u32x4*)(sb + idx + 4);
      bl.q[0] = *(const u32x4*)(sb + 2048 + idx);
      bl.q[1] = *(const u32x4*)(sb + 2048 + idx + 4);
#else
      size_t base = (((size_t)t * 8 + c) * 32 + lane) * 8;
      bh.q[0] = *(const u32x4*)(bhi + base);
      bh.q[1] = *(const u32x4*)(bhi + base + 4);
      bl.q[0] = *(const u32x4*)(blo + base);
      bl.q[1] = *(const u32x4*)(blo + base + 4);
#endif
      acc0 = __builtin_amdgcn_wmma_f32_16x16x32_bf16(false, a_hi[c].v, false, bh.v,
                                                     (short)0, acc0, false, false);
      acc1 = __builtin_amdgcn_wmma_f32_16x16x32_bf16(false, a_hi[c].v, false, bl.v,
                                                     (short)0, acc1, false, false);
      acc2 = __builtin_amdgcn_wmma_f32_16x16x32_bf16(false, a_lo[c].v, false, bh.v,
                                                     (short)0, acc2, false, false);
    }
    // C/D layout: lane n = lane&15, row M = v + 8*(lane>>4)
    const int kidx = t * 16 + (lane & 15);
    const float e2v = e2[kidx];
#pragma unroll
    for (int v = 0; v < 8; ++v) {
      float sim   = acc0[v] + acc1[v] + acc2[v];
      float score = e2v - 2.0f * sim;          // ||x||^2 constant per row: drop
      bool better = score < mins[v];
      mins[v] = better ? score : mins[v];
      mini[v] = better ? kidx  : mini[v];
    }
  }

  // ---- argmin reduce across the 16 lanes of each half (same rows) ----
#pragma unroll
  for (int m = 1; m <= 8; m <<= 1) {
#pragma unroll
    for (int v = 0; v < 8; ++v) {
      float os = __shfl_xor(mins[v], m, 32);
      int   oi = __shfl_xor(mini[v], m, 32);
      bool take = (os < mins[v]) || (os == mins[v] && oi < mini[v]);
      mins[v] = take ? os : mins[v];
      mini[v] = take ? oi : mini[v];
    }
  }
  // exchange halves so every lane knows all 16 row winners
  const int half = lane >> 4;
  int idxlo[8], idxhi[8];   // rows 0..7 / rows 8..15
#pragma unroll
  for (int v = 0; v < 8; ++v) {
    int other = __shfl_xor(mini[v], 16, 32);
    idxlo[v] = half ? other   : mini[v];
    idxhi[v] = half ? mini[v] : other;
  }

  // ---- cooperative gather of winning codes + output + loss ----
  float lsum = 0.f;
#pragma unroll
  for (int r = 0; r < 16; ++r) {
    const int row = rowbase + r;
    const int k   = (r < 8) ? idxlo[r] : idxhi[r - 8];
    const float4* qs = (const float4*)(eT + (size_t)k * D_DIM);
    const float4* xs = (const float4*)(x + (size_t)row * D_DIM);
    float4*       os = (float4*)(out + (size_t)row * D_DIM);
#pragma unroll
    for (int j = 0; j < 2; ++j) {
      int e = j * 32 + lane;            // 64 float4 per row
      float4 q  = qs[e];
      float4 xv = xs[e];
      os[e] = q;                        // straight-through value == q
      float dx = q.x - xv.x, dy = q.y - xv.y;
      float dz = q.z - xv.z, dw = q.w - xv.w;
      lsum += dx * dx + dy * dy + dz * dz + dw * dw;
    }
  }
#pragma unroll
  for (int m = 1; m < 32; m <<= 1) lsum += __shfl_xor(lsum, m, 32);

  __shared__ float wsum[8];
  if (lane == 0) wsum[wave] = lsum;
  __syncthreads();
  if (threadIdx.x == 0) {
    float s = 0.f;
#pragma unroll
    for (int w = 0; w < 8; ++w) s += wsum[w];
    partial[blockIdx.x] = s;
  }
}

// ---------------------------------------------------------------------------
// Final deterministic loss reduction: loss = (1 + BETA) * sum((q - x)^2)
// ---------------------------------------------------------------------------
__global__ void vq_reduce(const float* __restrict__ partial,
                          float* __restrict__ out_loss) {
  __shared__ float sm[256];
  sm[threadIdx.x] = partial[threadIdx.x];
  __syncthreads();
  for (int s = 128; s > 0; s >>= 1) {
    if (threadIdx.x < s) sm[threadIdx.x] += sm[threadIdx.x + s];
    __syncthreads();
  }
  if (threadIdx.x == 0) *out_loss = 1.25f * sm[0];
}

extern "C" void kernel_launch(void* const* d_in, const int* in_sizes, int n_in,
                              void* d_out, int out_size, void* d_ws, size_t ws_size,
                              hipStream_t stream) {
  (void)in_sizes; (void)n_in; (void)ws_size;
  const float* x   = (const float*)d_in[0];   // [32,32,32,256] f32
  const float* emb = (const float*)d_in[1];   // [256,1024] f32
  float* out = (float*)d_out;                 // 8388608 quantized + 1 loss

  char* ws = (char*)d_ws;
  uint32_t* bhi   = (uint32_t*)(ws + BHI_OFF);
  uint32_t* blo   = (uint32_t*)(ws + BLO_OFF);
  float*    eT    = (float*)(ws + ET_OFF);
  float*    e2    = (float*)(ws + E2_OFF);
  float*    part  = (float*)(ws + PART_OFF);

  vq_prep_e<<<4, 256, 0, stream>>>(emb, eT, e2);
  vq_pack_b<<<64, 256, 0, stream>>>(emb, bhi, blo);
  vq_main<<<N_ROWS / 128, 256, 0, stream>>>(x, eT, e2, bhi, blo, out, part);
  vq_reduce<<<1, 256, 0, stream>>>(part, out + (out_size - 1));
}
